// LatentGraphNetwork_25744033972839
// MI455X (gfx1250) — matrix operand order repaired and checked by reading, not compiled, not run on hardware
//
#include <hip/hip_runtime.h>

#define D 128
#define NROWS (64 * 4096)
#define NTILES (NROWS / 16)

typedef __attribute__((ext_vector_type(16))) _Float16 v16h;
typedef __attribute__((ext_vector_type(8)))  _Float16 v8h;
typedef __attribute__((ext_vector_type(8)))  float    v8f;

// ---------------- Prologue 1: W_ag = W_msg @ graph ; b_ag = b_msg @ graph ----
__global__ __launch_bounds__(128) void lgn_prep1(
    const float* __restrict__ Wmsg, const float* __restrict__ graph,
    const float* __restrict__ bmsg,
    float* __restrict__ Wag, float* __restrict__ bag)
{
  const int i = blockIdx.x, g = threadIdx.x;
  float s = 0.f;
  for (int h = 0; h < D; ++h) s += Wmsg[i * D + h] * graph[h * D + g];
  Wag[i * D + g] = s;
  if (i == 0) {
    float t = 0.f;
    for (int h = 0; h < D; ++h) t += bmsg[h] * graph[h * D + g];
    bag[g] = t;
  }
}

// ---- Prologue 2: W_eff = W_upd[:128] + W_ag @ W_upd[128:];  b_eff likewise.
// Store W_eff^T and W_out^T as fp16 (B-fragment friendly layout).
__global__ __launch_bounds__(128) void lgn_prep2(
    const float* __restrict__ Wupd, const float* __restrict__ bupd,
    const float* __restrict__ Wag,  const float* __restrict__ bag,
    const float* __restrict__ Wout,
    _Float16* __restrict__ WeffT, _Float16* __restrict__ WoutT,
    float* __restrict__ beff)
{
  const int i = blockIdx.x, o = threadIdx.x;
  float s = Wupd[i * D + o];
  for (int g = 0; g < D; ++g) s += Wag[i * D + g] * Wupd[(D + g) * D + o];
  WeffT[o * D + i] = (_Float16)s;
  WoutT[o * D + i] = (_Float16)Wout[i * D + o];
  if (i == 0) {
    float t = bupd[o];
    for (int g = 0; g < D; ++g) t += bag[g] * Wupd[(D + g) * D + o];
    beff[o] = t;
  }
}

// ---------------- Main fused kernel: y = LN(GELU(x@W_eff+b)@W_out+b) ---------
__global__ __launch_bounds__(256) void lgn_main(
    const float* __restrict__ x,
    const _Float16* __restrict__ WeffT, const _Float16* __restrict__ WoutT,
    const float* __restrict__ beff, const float* __restrict__ bout,
    const float* __restrict__ gamma, const float* __restrict__ beta,
    float* __restrict__ out)
{
  __shared__ __align__(32) _Float16 sWeff[D * D];     // 32 KB, [o][i]
  __shared__ __align__(32) _Float16 sWout[D * D];     // 32 KB, [o][i]
  __shared__ __align__(32) _Float16 sH[8][16 * D];    // 32 KB, per-wave 16x128
  __shared__ float sBeff[D], sBout[D], sGamma[D], sBeta[D];

  const int tid = threadIdx.x;

  // ---- stage weights/vectors into LDS (once per block) ----
  {
    const float4* se = (const float4*)WeffT;
    const float4* so = (const float4*)WoutT;
    float4* de = (float4*)sWeff;
    float4* dq = (float4*)sWout;
    for (int i = tid; i < (D * D) / 8; i += 256) { de[i] = se[i]; dq[i] = so[i]; }
    if (tid < D) {
      sBeff[tid] = beff[tid];  sBout[tid] = bout[tid];
      sGamma[tid] = gamma[tid]; sBeta[tid] = beta[tid];
    }
  }
  __syncthreads();

  const int wave = tid >> 5;
  const int lane = tid & 31;
  const int hlf  = lane >> 4;     // which 16-lane half
  const int ml   = lane & 15;

  const int tile = blockIdx.x * 8 + wave;          // grid sized so all valid
  const size_t rowBase = (size_t)tile * 16;
  const float* xrow = x + (rowBase + (size_t)ml) * D;

  float vBeff[8], vBout[8], vG[8], vBt[8];
#pragma unroll
  for (int nt = 0; nt < 8; ++nt) {
    const int col = nt * 16 + ml;
    vBeff[nt] = sBeff[col]; vBout[nt] = sBout[col];
    vG[nt]    = sGamma[col]; vBt[nt]  = sBeta[col];
  }

  // ---- build A fragments (16x32 fp16 layout: lane=row, two 8-elem runs) ----
  v16h a[4];
#pragma unroll
  for (int kc = 0; kc < 4; ++kc) {
    const int kb = kc * 32 + hlf * 8;
    float4 f0 = *(const float4*)(xrow + kb);
    float4 f1 = *(const float4*)(xrow + kb + 4);
    float4 f2 = *(const float4*)(xrow + kb + 16);
    float4 f3 = *(const float4*)(xrow + kb + 20);
    v16h av;
    av[0]=(_Float16)f0.x;  av[1]=(_Float16)f0.y;  av[2]=(_Float16)f0.z;  av[3]=(_Float16)f0.w;
    av[4]=(_Float16)f1.x;  av[5]=(_Float16)f1.y;  av[6]=(_Float16)f1.z;  av[7]=(_Float16)f1.w;
    av[8]=(_Float16)f2.x;  av[9]=(_Float16)f2.y;  av[10]=(_Float16)f2.z; av[11]=(_Float16)f2.w;
    av[12]=(_Float16)f3.x; av[13]=(_Float16)f3.y; av[14]=(_Float16)f3.z; av[15]=(_Float16)f3.w;
    a[kc] = av;
  }

  // ---- GEMM1 (x @ W_eff + b_eff), exact GELU, write fp16 H to LDS ----
  _Float16* Hw = sH[wave];
#pragma unroll
  for (int nt = 0; nt < 8; ++nt) {
    v8f c;
#pragma unroll
    for (int r = 0; r < 8; ++r) c[r] = vBeff[nt];
#pragma unroll
    for (int kc = 0; kc < 4; ++kc) {
      const v8h b0 = *(const v8h*)(sWeff + (nt * 16 + ml) * D + kc * 32 + hlf * 8);
      const v8h b1 = *(const v8h*)(sWeff + (nt * 16 + ml) * D + kc * 32 + hlf * 8 + 16);
      const v16h b = __builtin_shufflevector(b0, b1, 0,1,2,3,4,5,6,7,8,9,10,11,12,13,14,15);
      c = __builtin_amdgcn_wmma_f32_16x16x32_f16(false, a[kc], false, b,
                                                 (short)0, c, false, false);
    }
#pragma unroll
    for (int r = 0; r < 8; ++r) {
      const float v = c[r];
      const float g = 0.5f * v * (1.0f + erff(v * 0.70710678118654752f));
      Hw[(r + hlf * 8) * D + nt * 16 + ml] = (_Float16)g;   // D-layout -> row major
    }
  }
  __syncthreads();   // wave-internal cross-lane H handoff (block-uniform)

  // ---- reload H as A fragments for GEMM2 ----
  v16h a2[4];
#pragma unroll
  for (int kc = 0; kc < 4; ++kc) {
    const int kb = kc * 32 + hlf * 8;
    v8h h0 = *(const v8h*)(Hw + ml * D + kb);
    v8h h1 = *(const v8h*)(Hw + ml * D + kb + 16);
    v16h av;
#pragma unroll
    for (int j = 0; j < 8; ++j) { av[j] = h0[j]; av[j + 8] = h1[j]; }
    a2[kc] = av;
  }

  // ---- GEMM2 (h @ W_out + b_out) ----
  v8f o2[8];
#pragma unroll
  for (int nt = 0; nt < 8; ++nt) {
    v8f c;
#pragma unroll
    for (int r = 0; r < 8; ++r) c[r] = vBout[nt];
#pragma unroll
    for (int kc = 0; kc < 4; ++kc) {
      const v8h b0 = *(const v8h*)(sWout + (nt * 16 + ml) * D + kc * 32 + hlf * 8);
      const v8h b1 = *(const v8h*)(sWout + (nt * 16 + ml) * D + kc * 32 + hlf * 8 + 16);
      const v16h b = __builtin_shufflevector(b0, b1, 0,1,2,3,4,5,6,7,8,9,10,11,12,13,14,15);
      c = __builtin_amdgcn_wmma_f32_16x16x32_f16(false, a2[kc], false, b,
                                                 (short)0, c, false, false);
    }
    o2[nt] = c;
  }

  // ---- LayerNorm over the 128 columns of each row, then store fp32 ----
#pragma unroll
  for (int r = 0; r < 8; ++r) {
    float s = 0.f, s2 = 0.f;
#pragma unroll
    for (int nt = 0; nt < 8; ++nt) { const float v = o2[nt][r]; s += v; s2 += v * v; }
#pragma unroll
    for (int off = 1; off < 16; off <<= 1) {   // reduce within each 16-lane half
      s  += __shfl_xor(s,  off, 32);
      s2 += __shfl_xor(s2, off, 32);
    }
    const float mean = s * (1.0f / 128.0f);
    const float var  = s2 * (1.0f / 128.0f) - mean * mean;
    const float rstd = rsqrtf(var + 1e-5f);
    float* orow = out + (rowBase + (size_t)(r + hlf * 8)) * D;
#pragma unroll
    for (int nt = 0; nt < 8; ++nt) {
      orow[nt * 16 + ml] = (o2[nt][r] - mean) * rstd * vG[nt] + vBt[nt];
    }
  }
}

// ---------------- launcher ----------------
extern "C" void kernel_launch(void* const* d_in, const int* in_sizes, int n_in,
                              void* d_out, int out_size, void* d_ws, size_t ws_size,
                              hipStream_t stream)
{
  const float* x     = (const float*)d_in[0];
  const float* graph = (const float*)d_in[1];
  const float* Wmsg  = (const float*)d_in[2];
  const float* bmsg  = (const float*)d_in[3];
  const float* Wupd  = (const float*)d_in[4];
  const float* bupd  = (const float*)d_in[5];
  const float* Wout  = (const float*)d_in[6];
  const float* bout  = (const float*)d_in[7];
  const float* gamma = (const float*)d_in[8];
  const float* beta  = (const float*)d_in[9];
  float* out = (float*)d_out;

  char* ws = (char*)d_ws;
  _Float16* WeffT = (_Float16*)(ws);            // 32768 B
  _Float16* WoutT = (_Float16*)(ws + 32768);    // 32768 B
  float*    beff  = (float*)(ws + 65536);       //   512 B
  float*    Wag   = (float*)(ws + 66048);       // 65536 B
  float*    bag   = (float*)(ws + 131584);      //   512 B

  lgn_prep1<<<128, 128, 0, stream>>>(Wmsg, graph, bmsg, Wag, bag);
  lgn_prep2<<<128, 128, 0, stream>>>(Wupd, bupd, Wag, bag, Wout, WeffT, WoutT, beff);
  lgn_main<<<NTILES / 8, 256, 0, stream>>>(x, WeffT, WoutT, beff, bout, gamma, beta, out);
}
